// FP8Linear_10574209483058
// MI455X (gfx1250) — 8-bit matrix operand order repaired and checked by reading, compiled, not run on hardware
//
#include <hip/hip_runtime.h>
#include <hip/hip_bf16.h>
#include <stdint.h>

typedef __attribute__((ext_vector_type(16))) int   v16i_t;
typedef __attribute__((ext_vector_type(8)))  float v8f_t;
typedef int v4i_b128 __attribute__((__vector_size__(16)));

#if defined(__AMDGCN__) && __has_builtin(__builtin_amdgcn_global_load_async_to_lds_b128)
#define USE_ASYNC_LDS 1
#else
#define USE_ASYNC_LDS 0
#endif

__device__ __forceinline__ unsigned int f32_to_e4m3(float x) {
    unsigned int u = __float_as_uint(x);
    unsigned int s = (u >> 24) & 0x80u;
    unsigned int a = u & 0x7fffffffu;
    if (a >= 0x7f800000u) return s | 0x7fu;
    if (a > 0x43e00000u)  return s | 0x7eu;
    int e = (int)(a >> 23) - 127;
    unsigned int m = a & 0x7fffffu;
    if (e < -6) {
        int sh = 20 + (-6 - e);
        if (sh > 31) return s;
        unsigned int mm = m | 0x800000u;
        unsigned int q = mm >> sh;
        unsigned int rem = mm & ((1u << sh) - 1u);
        unsigned int half = 1u << (sh - 1);
        if (rem > half || (rem == half && (q & 1u))) q++;
        return s | q;
    }
    unsigned int q = m >> 20;
    unsigned int rem = m & 0xfffffu;
    if (rem > 0x80000u || (rem == 0x80000u && (q & 1u))) q++;
    unsigned int r = (((unsigned int)(e + 7)) << 3) + q;
    if (r > 0x7eu) r = 0x7eu;
    return s | r;
}

__device__ __forceinline__ unsigned int pack4_e4m3(float4 f, float sinv) {
    return  f32_to_e4m3(f.x * sinv)
         | (f32_to_e4m3(f.y * sinv) << 8)
         | (f32_to_e4m3(f.z * sinv) << 16)
         | (f32_to_e4m3(f.w * sinv) << 24);
}

__global__ __launch_bounds__(256) void quant_act(const float* __restrict__ x,
                                                 const float* __restrict__ scale,
                                                 uint4* __restrict__ q,
                                                 long long nvec16) {
    long long i = (long long)blockIdx.x * 256 + threadIdx.x;
    if (i >= nvec16) return;
    float sinv = 1.0f / scale[0];
    const float4* xv = (const float4*)x + i * 4;
    float4 f0 = xv[0], f1 = xv[1], f2 = xv[2], f3 = xv[3];
    uint4 r;
    r.x = pack4_e4m3(f0, sinv);
    r.y = pack4_e4m3(f1, sinv);
    r.z = pack4_e4m3(f2, sinv);
    r.w = pack4_e4m3(f3, sinv);
    q[i] = r;
}

__global__ __launch_bounds__(1024) void quant_wt_t(const float* __restrict__ w,
                                                   const float* __restrict__ scale,
                                                   unsigned char* __restrict__ wt,
                                                   int K, int OUT) {
    __shared__ float tile[32][33];
    float sinv = 1.0f / scale[0];
    int k0 = blockIdx.x * 32;
    int o0 = blockIdx.y * 32;
    int tx = threadIdx.x, ty = threadIdx.y;
    tile[ty][tx] = w[(size_t)(k0 + ty) * OUT + (o0 + tx)];
    __syncthreads();
    wt[(size_t)(o0 + ty) * K + (k0 + tx)] =
        (unsigned char)f32_to_e4m3(tile[tx][ty] * sinv);
}

#define BM 128
#define BN 256
#define BK 128

__device__ __forceinline__ void stage_tiles(const unsigned char* __restrict__ A,
                                            const unsigned char* __restrict__ Bt,
                                            int K, int tid, int m0, int n0, int kb,
                                            unsigned char* sA, unsigned char* sB) {
    const unsigned char* gA = A  + (size_t)m0 * K + kb;
    const unsigned char* gB = Bt + (size_t)n0 * K + kb;
#if USE_ASYNC_LDS
#pragma unroll
    for (int i = 0; i < 4; ++i) {
        int o = i * 4096 + tid * 16;
        int r = o >> 7, c = o & (BK - 1);
        __builtin_amdgcn_global_load_async_to_lds_b128(
            (v4i_b128*)(gA + (size_t)r * K + c), (v4i_b128*)(sA + o), 0, 0);
    }
#pragma unroll
    for (int i = 0; i < 8; ++i) {
        int o = i * 4096 + tid * 16;
        int r = o >> 7, c = o & (BK - 1);
        __builtin_amdgcn_global_load_async_to_lds_b128(
            (v4i_b128*)(gB + (size_t)r * K + c), (v4i_b128*)(sB + o), 0, 0);
    }
#else
#pragma unroll
    for (int i = 0; i < 4; ++i) {
        int o = i * 4096 + tid * 16;
        int r = o >> 7, c = o & (BK - 1);
        *(uint4*)(sA + o) = *(const uint4*)(gA + (size_t)r * K + c);
    }
#pragma unroll
    for (int i = 0; i < 8; ++i) {
        int o = i * 4096 + tid * 16;
        int r = o >> 7, c = o & (BK - 1);
        *(uint4*)(sB + o) = *(const uint4*)(gB + (size_t)r * K + c);
    }
#endif
}

__device__ __forceinline__ void wait_async_le12() {
#if USE_ASYNC_LDS
#if __has_builtin(__builtin_amdgcn_s_wait_asynccnt)
    __builtin_amdgcn_s_wait_asynccnt(12);
#else
    asm volatile("s_wait_asynccnt 0xc" ::: "memory");
#endif
#endif
}
__device__ __forceinline__ void wait_async_zero() {
#if USE_ASYNC_LDS
#if __has_builtin(__builtin_amdgcn_s_wait_asynccnt)
    __builtin_amdgcn_s_wait_asynccnt(0);
#else
    asm volatile("s_wait_asynccnt 0x0" ::: "memory");
#endif
#endif
}

__device__ __forceinline__ void compute_step(const unsigned char* __restrict__ At,
                                             const unsigned char* __restrict__ Bb,
                                             int wm, int wn, int half, int l16,
                                             v8f_t (&acc)[4][4]) {
    v16i_t af[4];
#pragma unroll
    for (int i = 0; i < 4; ++i) {
        const unsigned int* row =
            (const unsigned int*)(At + (wm + i * 16 + l16) * BK);
#pragma unroll
        for (int v = 0; v < 16; ++v) {
            int k0 = ((v >> 3) * 64) + (((v >> 1) & 3) * 16)
                   + ((v & 1) * 4) + half * 8;
            af[i][v] = (int)row[k0 >> 2];
        }
    }
#pragma unroll
    for (int j = 0; j < 4; ++j) {
        const unsigned int* row =
            (const unsigned int*)(Bb + (wn + j * 16 + l16) * BK);
        v16i_t bf;
#pragma unroll
        for (int v = 0; v < 16; ++v) {
            int k0 = ((v >> 3) * 64) + (((v >> 1) & 3) * 16)
                   + ((v & 1) * 4) + half * 8;
            bf[v] = (int)row[k0 >> 2];
        }
#pragma unroll
        for (int i = 0; i < 4; ++i)
            acc[i][j] = __builtin_amdgcn_wmma_f32_16x16x128_fp8_fp8(
                af[i], bf, (short)0, acc[i][j], false, false);
    }
}

__global__ __launch_bounds__(256) void fp8_gemm(const unsigned char* __restrict__ A,
                                                const unsigned char* __restrict__ Bt,
                                                const float* __restrict__ bias,
                                                const float* __restrict__ s_in,
                                                const float* __restrict__ s_w,
                                                float* __restrict__ out,
                                                int M, int N, int K) {
    __shared__ unsigned char Abuf[2][BM * BK];
    __shared__ unsigned char Bbuf[2][BN * BK];

    const int tid  = threadIdx.x;
    const int wave = tid >> 5;
    const int lane = tid & 31;
    const int half = lane >> 4;
    const int l16  = lane & 15;

    const int m0 = blockIdx.y * BM;
    const int n0 = blockIdx.x * BN;
    const int wm = (wave >> 2) * 64;
    const int wn = (wave & 3) * 64;

    v8f_t acc[4][4] = {};

    const int nk = K / BK;

    stage_tiles(A, Bt, K, tid, m0, n0, 0, Abuf[0], Bbuf[0]);

    for (int t = 0; t < nk - 1; ++t) {
        const int cur = t & 1;
        stage_tiles(A, Bt, K, tid, m0, n0, (t + 1) * BK,
                    Abuf[cur ^ 1], Bbuf[cur ^ 1]);
        wait_async_le12();
        __syncthreads();
        compute_step(Abuf[cur], Bbuf[cur], wm, wn, half, l16, acc);
        __syncthreads();
    }
    wait_async_zero();
    __syncthreads();
    compute_step(Abuf[(nk - 1) & 1], Bbuf[(nk - 1) & 1], wm, wn, half, l16, acc);

    const float sc2 = s_in[0] * s_w[0];
#pragma unroll
    for (int i = 0; i < 4; ++i) {
#pragma unroll
        for (int j = 0; j < 4; ++j) {
            const int ncol = n0 + wn + j * 16 + l16;
            const float bv = bias[ncol];
            const int mbase = m0 + wm + i * 16 + half * 8;
#pragma unroll
            for (int r = 0; r < 8; ++r) {
                out[(size_t)(mbase + r) * N + ncol] = acc[i][j][r] * sc2 + bv;
            }
        }
    }
}

extern "C" void kernel_launch(void* const* d_in, const int* in_sizes, int n_in,
                              void* d_out, int out_size, void* d_ws, size_t ws_size,
                              hipStream_t stream) {
    const float* inp    = (const float*)d_in[0];
    const float* weight = (const float*)d_in[1];
    const float* bias   = (const float*)d_in[2];
    const float* s_in   = (const float*)d_in[3];
    const float* s_w    = (const float*)d_in[4];
    float* out = (float*)d_out;

    const int OUT = in_sizes[2];
    const int K   = in_sizes[1] / OUT;
    const long long Mll = (long long)in_sizes[0] / K;
    const int M = (int)Mll;

    unsigned char* xq  = (unsigned char*)d_ws;
    unsigned char* wqt = xq + (size_t)M * K;

    long long nvec16 = ((long long)M * K) / 16;
    long long qblocks = (nvec16 + 255) / 256;
    quant_act<<<(unsigned int)qblocks, 256, 0, stream>>>(inp, s_in, (uint4*)xq, nvec16);

    dim3 tb(32, 32), tg(K / 32, OUT / 32);
    quant_wt_t<<<tg, tb, 0, stream>>>(weight, s_w, wqt, K, OUT);

    dim3 gb(256), gg(OUT / BN, M / BM);
    fp8_gemm<<<gg, gb, 0, stream>>>(xq, wqt, bias, s_in, s_w, out, M, OUT, K);
}
